// HGTLayer_6665789243663
// MI455X (gfx1250) — matrix operand order repaired and checked by reading, not compiled, not run on hardware
//
#include <hip/hip_runtime.h>
#include <hip/hip_bf16.h>
#include <math.h>

// ---------------------------------------------------------------------------
// HGT layer for MI455X (gfx1250).
// Dense GEMMs via v_wmma_f32_16x16x32_f16 (f16 in, f32 accumulate), with
// B tiles staged into LDS by GLOBAL_LOAD_ASYNC_TO_LDS_B128 (ASYNCcnt-tracked
// DMA, double-buffered, one barrier per k-step) and A fragments register-
// pipelined. Per-relation 64x64 head transforms are folded into the
// projection weights; edge softmax/agg is VALU + float atomics.
// ---------------------------------------------------------------------------

#define NA   40000
#define NP_  60000
#define NE   80000
#define DMODEL 512
#define NH   8
#define DK   64

typedef __attribute__((ext_vector_type(16))) _Float16 v16h;
typedef __attribute__((ext_vector_type(8)))  float    v8f;
typedef __attribute__((ext_vector_type(4)))  int      v4i;

#if defined(__has_builtin)
#if __has_builtin(__builtin_amdgcn_global_load_async_to_lds_b128) && \
    __has_builtin(__builtin_amdgcn_s_wait_asynccnt)
#define USE_ASYNC_LDS 1
#endif
#endif
#ifndef USE_ASYNC_LDS
#define USE_ASYNC_LDS 0
#endif

#if USE_ASYNC_LDS
// builtin signature (from hipcc diagnostic): (v4i AS1*, v4i AS3*, imm, imm)
#define GLOBAL_V4I(p) ((__attribute__((address_space(1))) v4i*)(v4i*)(p))
#define LDS_V4I(p)    ((__attribute__((address_space(3))) v4i*)(v4i*)(p))
#endif

// ---------------- WMMA fragment packing -------------------------------------
// A-matrix 16x32 f16 (M x K), wave32 layout (ISA 7.12.2):
//   lane = (K-half << 4) | (M & 15); within a lane, 8 dwords = 16 halfs.
//   kk = K%32: half = (kk&8)?1:0 ; vgpr v = ((kk>>4)<<2) + ((kk&7)>>1)
__device__ __forceinline__ size_t pack_index_A(int row, int k, int K) {
  int mtile = row >> 4;
  int ktile = k >> 5;
  int kk    = k & 31;
  int half  = (kk & 8) ? 1 : 0;
  int v     = ((kk >> 4) << 2) + ((kk & 7) >> 1);
  int slot  = v * 2 + (kk & 1);
  int lane  = (half << 4) + (row & 15);
  int kt    = K >> 5;
  return (((size_t)(mtile * kt + ktile)) * 32 + lane) * 16 + slot;
}

// B-matrix 32x16 f16 (K x N): same K order as A: half = (kk>>3)&1,
// slot = ((kk>>4)<<3) + (kk&7); lane = (half<<4) + N.
__device__ __forceinline__ size_t pack_index_B(int k, int n, int K) {
  int ntile = n >> 4;
  int ktile = k >> 5;
  int kk    = k & 31;
  int half  = (kk >> 3) & 1;
  int v     = ((kk >> 4) << 2) + ((kk & 7) >> 1);
  int slot  = v * 2 + (kk & 1);
  int lane  = (half << 4) + (n & 15);
  int kt    = K >> 5;
  return (((size_t)(ntile * kt + ktile)) * 32 + lane) * 16 + slot;
}

// ---------------- small utility kernels -------------------------------------
__global__ void fill_f32(float* p, size_t n, float v) {
  size_t i = (size_t)blockIdx.x * blockDim.x + threadIdx.x;
  if (i < n) p[i] = v;
}

// f32 row-major (rows x K) -> packed f16 A fragments (with scale)
__global__ void pack_a_f32(const float* __restrict__ src, _Float16* __restrict__ dst,
                           int rows, int K, float scale) {
  size_t i = (size_t)blockIdx.x * blockDim.x + threadIdx.x;
  size_t total = (size_t)rows * K;
  if (i >= total) return;
  int row = (int)(i / K);
  int k   = (int)(i % K);
  dst[pack_index_A(row, k, K)] = (_Float16)(src[i] * scale);
}

// W' [i][h*64+e] = sum_d W[i][h*64+d] * att[h][d][e]  -> packed B f16
__global__ void fold_pack_w(const float* __restrict__ W,   // 512x512
                            const float* __restrict__ att, // H x 64 x 64
                            _Float16* __restrict__ dstPack, int colOff) {
  int idx = blockIdx.x * blockDim.x + threadIdx.x;
  if (idx >= DMODEL * DMODEL) return;
  int i = idx >> 9;
  int c = idx & 511;
  int h = c >> 6;
  int e = c & 63;
  const float* wrow = W + (size_t)i * DMODEL + h * DK;
  const float* a    = att + (size_t)h * DK * DK + e;
  float s = 0.f;
  #pragma unroll 8
  for (int d = 0; d < DK; ++d) s += wrow[d] * a[d * DK];
  dstPack[pack_index_B(i, colOff + c, DMODEL)] = (_Float16)s;
}

// plain W (512x512) -> packed B f16
__global__ void pack_w(const float* __restrict__ W, _Float16* __restrict__ dstPack, int colOff) {
  int idx = blockIdx.x * blockDim.x + threadIdx.x;
  if (idx >= DMODEL * DMODEL) return;
  int i = idx >> 9;
  int c = idx & 511;
  dstPack[pack_index_B(i, colOff + c, DMODEL)] = (_Float16)W[(size_t)i * DMODEL + c];
}

// b'[c] = sum_d b[h*64+d] * att[h][d][e]
__global__ void fold_bias(const float* __restrict__ b, const float* __restrict__ att,
                          float* __restrict__ dst, int off) {
  int c = blockIdx.x * blockDim.x + threadIdx.x;
  if (c >= DMODEL) return;
  int h = c >> 6, e = c & 63;
  const float* a = att + (size_t)h * DK * DK + e;
  float s = 0.f;
  for (int d = 0; d < DK; ++d) s += b[h * DK + d] * a[d * DK];
  dst[off + c] = s;
}

__global__ void copy_bias(const float* __restrict__ b, float* __restrict__ dst, int off) {
  int c = blockIdx.x * blockDim.x + threadIdx.x;
  if (c < DMODEL) dst[off + c] = b[c];
}

// ---------------- WMMA GEMM with async-LDS-staged B --------------------------
// Block = 256 threads = 8 waves. Block computes a 128x64 output tile:
// wave w -> m-tile (mblk*8+w); all waves share 4 n-tiles of B, staged in LDS
// (double-buffered 2 x 4KB). Per k-step:
//   s_wait_asynccnt 0  (buffer for kt fully landed)
//   barrier            (also guarantees everyone is done reading buf^1)
//   async-fill buf^1 for kt+1   -- DMA overlaps the WMMAs below
//   ds_load fragments of buf, 4x v_wmma, A fragment register-pipelined.
// mode 0:  out[row*Ntot+n] = acc + bias[n]
// mode 1:  out[row*512 +n] = (acc + bias[n])*sigmoid(skip[t]) + h[row*512+n]*(1-sigmoid)
__global__ void __launch_bounds__(256)
gemm_wmma_f16(const _Float16* __restrict__ Apack,
              const _Float16* __restrict__ Bpack,
              int Mtiles, int Ntot, int K,
              const float* __restrict__ bias,
              float* __restrict__ out,
              const float* __restrict__ horig,
              const float* __restrict__ skip,
              int typeIdx, int mode) {
  // 2 buffers x 4 n-tiles x (32 lanes x 16 halfs) = 2 x 4KB
  __shared__ __align__(32) _Float16 smem[2 * 4 * 32 * 16];

  int NQ   = (Ntot >> 4) >> 2;          // groups of 4 n-tiles
  int mblk = blockIdx.x / NQ;
  int nq   = blockIdx.x % NQ;
  int wave = threadIdx.x >> 5;
  int lane = threadIdx.x & 31;
  int ktiles = K >> 5;

  int  mt     = mblk * 8 + wave;
  bool active = (mt < Mtiles);
  int  mtc    = active ? mt : (Mtiles - 1);   // clamp: keep barriers uniform

  const _Float16* aBase = Apack + (((size_t)mtc * ktiles) * 32 + lane) * 16;

  // cooperative B staging: thread t moves one 16B chunk; t = j*64 + chunk
  int jt = threadIdx.x >> 6;            // which of the 4 n-tiles
  const float4* bSrc = (const float4*)(Bpack + ((size_t)(nq * 4 + jt) * ktiles) * (32 * 16))
                       + (threadIdx.x & 63);
  float4* sm4 = (float4*)smem;          // [2][256] float4 chunks

  v16h a_cur = *(const v16h*)aBase;     // register-pipelined A fragment (kt=0)
  v8f acc0 = {}, acc1 = {}, acc2 = {}, acc3 = {};

#if USE_ASYNC_LDS
  // prologue: async fill of buffer 0 with kt=0
  __builtin_amdgcn_global_load_async_to_lds_b128(
      GLOBAL_V4I(bSrc), LDS_V4I(sm4 + threadIdx.x), 0, 0);
#else
  float4 breg = bSrc[0];                // prefetch kt = 0 through registers
#endif

  for (int kt = 0; kt < ktiles; ++kt) {
    int buf = kt & 1;
#if USE_ASYNC_LDS
    __builtin_amdgcn_s_wait_asynccnt(0);       // fill(kt) landed in LDS
    __syncthreads();                           // + nobody still reads buf^1
    if (kt + 1 < ktiles)                       // DMA next tile during compute
      __builtin_amdgcn_global_load_async_to_lds_b128(
          GLOBAL_V4I(bSrc + (size_t)(kt + 1) * 64),
          LDS_V4I(sm4 + (buf ^ 1) * 256 + threadIdx.x), 0, 0);
#else
    sm4[buf * 256 + threadIdx.x] = breg;       // ds_store_b128
    __syncthreads();
    if (kt + 1 < ktiles) breg = bSrc[(size_t)(kt + 1) * 64];
#endif

    v16h a = a_cur;
    if (kt + 1 < ktiles)                       // pipeline next A fragment
      a_cur = *(const v16h*)(aBase + (size_t)(kt + 1) * (32 * 16));

    const v16h* bf = (const v16h*)(smem + (size_t)buf * (4 * 32 * 16));
    v16h b0 = bf[0 * 32 + lane];
    v16h b1 = bf[1 * 32 + lane];
    v16h b2 = bf[2 * 32 + lane];
    v16h b3 = bf[3 * 32 + lane];
    acc0 = __builtin_amdgcn_wmma_f32_16x16x32_f16(false, a, false, b0, (short)0, acc0, false, false);
    acc1 = __builtin_amdgcn_wmma_f32_16x16x32_f16(false, a, false, b1, (short)0, acc1, false, false);
    acc2 = __builtin_amdgcn_wmma_f32_16x16x32_f16(false, a, false, b2, (short)0, acc2, false, false);
    acc3 = __builtin_amdgcn_wmma_f32_16x16x32_f16(false, a, false, b3, (short)0, acc3, false, false);
    // single barrier per iteration is sufficient with double buffering:
    // any wave reaches barrier(kt+1) only after its compute(kt), so the
    // buffer write at kt+2 (after barrier(kt+1)) cannot race reads at kt.
  }

  if (!active) return;

  float alpha = 1.f, beta = 0.f;
  if (mode == 1) {
    float sk = skip[typeIdx];
    alpha = 1.f / (1.f + __expf(-sk));
    beta = 1.f - alpha;
  }
  int col0 = lane & 15;
  int rowHalf = (lane >> 4) * 8;   // C layout: vgpr r -> row = r + 8*(lane/16)
  v8f accs[4] = {acc0, acc1, acc2, acc3};
  #pragma unroll
  for (int j = 0; j < 4; ++j) {
    int n = (nq * 4 + j) * 16 + col0;
    float bn = bias[n];
    #pragma unroll
    for (int r = 0; r < 8; ++r) {
      int row = mt * 16 + rowHalf + r;
      float val = accs[j][r];
      if (mode == 0) {
        out[(size_t)row * Ntot + n] = val + bn;
      } else {
        out[(size_t)row * DMODEL + n] =
            (val + bn) * alpha + horig[(size_t)row * DMODEL + n] * beta;
      }
    }
  }
}

// ---------------- edge softmax / aggregation ---------------------------------
__device__ __forceinline__ void atomicMaxF(float* addr, float v) {
  if (v >= 0.f) atomicMax((int*)addr, __float_as_int(v));
  else          atomicMin((unsigned int*)addr, (unsigned int)__float_as_int(v));
}

__global__ void edge_score(const float* __restrict__ Kbuf, int ldk, int koff,
                           const float* __restrict__ Qbuf, int ldq, int qoff,
                           const int* __restrict__ src, const int* __restrict__ dst,
                           const float* __restrict__ pri, int prioff,
                           float* __restrict__ sbuf, float* __restrict__ maxbuf) {
  int i = blockIdx.x * blockDim.x + threadIdx.x;
  if (i >= NE * NH) return;
  int e = i >> 3, h = i & 7;
  const float4* kv = (const float4*)(Kbuf + (size_t)src[e] * ldk + koff + h * DK);
  const float4* qv = (const float4*)(Qbuf + (size_t)dst[e] * ldq + qoff + h * DK);
  float s = 0.f;
  #pragma unroll
  for (int t = 0; t < DK / 4; ++t) {
    float4 a = kv[t], b = qv[t];
    s += a.x * b.x + a.y * b.y + a.z * b.z + a.w * b.w;
  }
  s *= pri[prioff + h] * 0.125f;  // 1/sqrt(64)
  sbuf[i] = s;
  atomicMaxF(maxbuf + (size_t)dst[e] * NH + h, s);
}

__global__ void edge_exp(const float* __restrict__ maxbuf, const int* __restrict__ dst,
                         float* __restrict__ sbuf, float* __restrict__ den) {
  int i = blockIdx.x * blockDim.x + threadIdx.x;
  if (i >= NE * NH) return;
  int e = i >> 3, h = i & 7;
  float ex = __expf(sbuf[i] - maxbuf[(size_t)dst[e] * NH + h]);
  sbuf[i] = ex;
  atomicAdd(den + (size_t)dst[e] * NH + h, ex);
}

__global__ void edge_agg(const float* __restrict__ Vbuf, int ldv, int voff,
                         const int* __restrict__ src, const int* __restrict__ dst,
                         const float* __restrict__ sbuf, const float* __restrict__ den,
                         float* __restrict__ t) {
  size_t i = (size_t)blockIdx.x * blockDim.x + threadIdx.x;
  if (i >= (size_t)NE * DMODEL) return;
  int e = (int)(i >> 9);
  int c = (int)(i & 511);
  int h = c >> 6;
  int d = dst[e];
  float a = sbuf[(size_t)e * NH + h] / den[(size_t)d * NH + h];
  atomicAdd(t + (size_t)d * DMODEL + c, Vbuf[(size_t)src[e] * ldv + voff + c] * a);
}

// ---------------- host side --------------------------------------------------
static inline int cdiv(long long a, long long b) { return (int)((a + b - 1) / b); }

extern "C" void kernel_launch(void* const* d_in, const int* in_sizes, int n_in,
                              void* d_out, int out_size, void* d_ws, size_t ws_size,
                              hipStream_t stream) {
  (void)in_sizes; (void)n_in; (void)out_size; (void)ws_size;

  const float* h_author   = (const float*)d_in[0];
  const float* h_paper    = (const float*)d_in[1];
  const int*   writes_src = (const int*)d_in[2];
  const int*   writes_dst = (const int*)d_in[3];
  const int*   cites_src  = (const int*)d_in[4];
  const int*   cites_dst  = (const int*)d_in[5];
  const int*   wb_src     = (const int*)d_in[6];
  const int*   wb_dst     = (const int*)d_in[7];
  const float* Wk = (const float*)d_in[8];
  const float* bk = (const float*)d_in[9];
  const float* Wq = (const float*)d_in[10];
  const float* bq = (const float*)d_in[11];
  const float* Wv = (const float*)d_in[12];
  const float* bv = (const float*)d_in[13];
  const float* Wa = (const float*)d_in[14];
  const float* ba = (const float*)d_in[15];
  const float* rel_att = (const float*)d_in[16];
  const float* rel_msg = (const float*)d_in[17];
  const float* rel_pri = (const float*)d_in[18];
  const float* skip    = (const float*)d_in[19];
  float* out = (float*)d_out;

  // workspace carve-out
  char* wsp = (char*)d_ws;
  size_t cur = 0;
  auto alloc = [&](size_t bytes) -> void* {
    void* p = wsp + cur;
    cur = (cur + bytes + 255) & ~(size_t)255;
    return p;
  };
  const size_t WMAT = (size_t)DMODEL * DMODEL;
  _Float16* packA_a  = (_Float16*)alloc((size_t)NA  * DMODEL * 2);
  _Float16* packA_p  = (_Float16*)alloc((size_t)NP_ * DMODEL * 2);
  _Float16* packW_a  = (_Float16*)alloc((size_t)DMODEL * 1536 * 2);
  _Float16* packW_p  = (_Float16*)alloc((size_t)DMODEL * 2560 * 2);
  _Float16* packWa0  = (_Float16*)alloc(WMAT * 2);
  _Float16* packWa1  = (_Float16*)alloc(WMAT * 2);
  float*    bias_a   = (float*)alloc(1536 * 4);
  float*    bias_p   = (float*)alloc(2560 * 4);
  float*    buf_a    = (float*)alloc((size_t)NA  * 1536 * 4);  // [K0|V0|Qa]
  float*    buf_p    = (float*)alloc((size_t)NP_ * 2560 * 4);  // [K1|V1|K2|V2|Qp]
  float*    t_a      = (float*)alloc((size_t)NA  * DMODEL * 4);
  float*    t_p      = (float*)alloc((size_t)NP_ * DMODEL * 4);
  _Float16* packT_a  = (_Float16*)alloc((size_t)NA  * DMODEL * 2);
  _Float16* packT_p  = (_Float16*)alloc((size_t)NP_ * DMODEL * 2);
  float*    maxAll   = (float*)alloc((size_t)(NP_ + NP_ + NA) * NH * 4);
  float*    denAll   = (float*)alloc((size_t)(NP_ + NP_ + NA) * NH * 4);
  float*    sbAll    = (float*)alloc((size_t)3 * NE * NH * 4);
  float* maxW = maxAll, *maxC = maxAll + (size_t)NP_ * NH, *maxB = maxAll + (size_t)2 * NP_ * NH;
  float* denW = denAll, *denC = denAll + (size_t)NP_ * NH, *denB = denAll + (size_t)2 * NP_ * NH;
  float* sb0 = sbAll, *sb1 = sbAll + (size_t)NE * NH, *sb2 = sbAll + (size_t)2 * NE * NH;

  const int TB = 256;

  // 0) init accumulators
  {
    size_t n = (size_t)(NP_ + NP_ + NA) * NH;
    fill_f32<<<cdiv(n, TB), TB, 0, stream>>>(maxAll, n, -INFINITY);
    fill_f32<<<cdiv(n, TB), TB, 0, stream>>>(denAll, n, 0.f);
    size_t nt = (size_t)NA * DMODEL;
    fill_f32<<<cdiv(nt, TB), TB, 0, stream>>>(t_a, nt, 0.f);
    size_t np = (size_t)NP_ * DMODEL;
    fill_f32<<<cdiv(np, TB), TB, 0, stream>>>(t_p, np, 0.f);
  }

  // 1) pack node features to f16 A fragments
  pack_a_f32<<<cdiv((size_t)NA * DMODEL, TB), TB, 0, stream>>>(h_author, packA_a, NA, DMODEL, 1.f);
  pack_a_f32<<<cdiv((size_t)NP_ * DMODEL, TB), TB, 0, stream>>>(h_paper, packA_p, NP_, DMODEL, 1.f);

  // 2) fold relation transforms into projection weights, pack to B fragments
  const int WG = cdiv(DMODEL * DMODEL, TB);
  const size_t REL = (size_t)NH * DK * DK;
  // authors: [Wk0*att0 | Wv0*msg0 | Wq0]
  fold_pack_w<<<WG, TB, 0, stream>>>(Wk,        rel_att,         packW_a, 0);
  fold_pack_w<<<WG, TB, 0, stream>>>(Wv,        rel_msg,         packW_a, 512);
  pack_w    <<<WG, TB, 0, stream>>>(Wq,                          packW_a, 1024);
  // papers: [Wk1*att1 | Wv1*msg1 | Wk1*att2 | Wv1*msg2 | Wq1]
  fold_pack_w<<<WG, TB, 0, stream>>>(Wk + WMAT, rel_att + REL,   packW_p, 0);
  fold_pack_w<<<WG, TB, 0, stream>>>(Wv + WMAT, rel_msg + REL,   packW_p, 512);
  fold_pack_w<<<WG, TB, 0, stream>>>(Wk + WMAT, rel_att + 2*REL, packW_p, 1024);
  fold_pack_w<<<WG, TB, 0, stream>>>(Wv + WMAT, rel_msg + 2*REL, packW_p, 1536);
  pack_w    <<<WG, TB, 0, stream>>>(Wq + WMAT,                   packW_p, 2048);
  // output weights
  pack_w<<<WG, TB, 0, stream>>>(Wa,        packWa0, 0);
  pack_w<<<WG, TB, 0, stream>>>(Wa + WMAT, packWa1, 0);
  // biases
  fold_bias<<<cdiv(DMODEL, TB), TB, 0, stream>>>(bk,          rel_att,         bias_a, 0);
  fold_bias<<<cdiv(DMODEL, TB), TB, 0, stream>>>(bv,          rel_msg,         bias_a, 512);
  copy_bias<<<cdiv(DMODEL, TB), TB, 0, stream>>>(bq,                           bias_a, 1024);
  fold_bias<<<cdiv(DMODEL, TB), TB, 0, stream>>>(bk + DMODEL, rel_att + REL,   bias_p, 0);
  fold_bias<<<cdiv(DMODEL, TB), TB, 0, stream>>>(bv + DMODEL, rel_msg + REL,   bias_p, 512);
  fold_bias<<<cdiv(DMODEL, TB), TB, 0, stream>>>(bk + DMODEL, rel_att + 2*REL, bias_p, 1024);
  fold_bias<<<cdiv(DMODEL, TB), TB, 0, stream>>>(bv + DMODEL, rel_msg + 2*REL, bias_p, 1536);
  copy_bias<<<cdiv(DMODEL, TB), TB, 0, stream>>>(bq + DMODEL,                  bias_p, 2048);

  // 3) fused projection GEMMs (WMMA, async-LDS-staged B)
  {
    int grid_a = cdiv(NA / 16, 8) * (1536 / 64);
    gemm_wmma_f16<<<grid_a, TB, 0, stream>>>(
        packA_a, packW_a, NA / 16, 1536, DMODEL, bias_a, buf_a, nullptr, nullptr, 0, 0);
    int grid_p = cdiv(NP_ / 16, 8) * (2560 / 64);
    gemm_wmma_f16<<<grid_p, TB, 0, stream>>>(
        packA_p, packW_p, NP_ / 16, 2560, DMODEL, bias_p, buf_p, nullptr, nullptr, 0, 0);
  }

  // 4) edge attention, one relation at a time
  const int EG = cdiv((size_t)NE * NH, TB);
  const int AG = cdiv((size_t)NE * DMODEL, TB);
  // relation 0: writes, author -> paper  (K0@buf_a+0, V0@buf_a+512, Qp@buf_p+2048)
  edge_score<<<EG, TB, 0, stream>>>(buf_a, 1536, 0,    buf_p, 2560, 2048,
                                    writes_src, writes_dst, rel_pri, 0, sb0, maxW);
  edge_exp  <<<EG, TB, 0, stream>>>(maxW, writes_dst, sb0, denW);
  edge_agg  <<<AG, TB, 0, stream>>>(buf_a, 1536, 512, writes_src, writes_dst, sb0, denW, t_p);
  // relation 1: cites, paper -> paper    (K1@buf_p+0, V1@buf_p+512, Qp@buf_p+2048)
  edge_score<<<EG, TB, 0, stream>>>(buf_p, 2560, 0,    buf_p, 2560, 2048,
                                    cites_src, cites_dst, rel_pri, NH, sb1, maxC);
  edge_exp  <<<EG, TB, 0, stream>>>(maxC, cites_dst, sb1, denC);
  edge_agg  <<<AG, TB, 0, stream>>>(buf_p, 2560, 512, cites_src, cites_dst, sb1, denC, t_p);
  // relation 2: written_by, paper -> author (K2@buf_p+1024, V2@buf_p+1536, Qa@buf_a+1024)
  edge_score<<<EG, TB, 0, stream>>>(buf_p, 2560, 1024, buf_a, 1536, 1024,
                                    wb_src, wb_dst, rel_pri, 2 * NH, sb2, maxB);
  edge_exp  <<<EG, TB, 0, stream>>>(maxB, wb_dst, sb2, denB);
  edge_agg  <<<AG, TB, 0, stream>>>(buf_p, 2560, 1536, wb_src, wb_dst, sb2, denB, t_a);

  // 5) pack aggregated messages (paper gets the 0.5 mean factor)
  pack_a_f32<<<cdiv((size_t)NA * DMODEL, TB), TB, 0, stream>>>(t_a, packT_a, NA, DMODEL, 1.f);
  pack_a_f32<<<cdiv((size_t)NP_ * DMODEL, TB), TB, 0, stream>>>(t_p, packT_p, NP_, DMODEL, 0.5f);

  // 6) output GEMMs fused with sigmoid-skip residual, straight into d_out
  {
    int grid_a = cdiv(NA / 16, 8) * (DMODEL / 64);
    gemm_wmma_f16<<<grid_a, TB, 0, stream>>>(
        packT_a, packWa0, NA / 16, DMODEL, DMODEL, ba, out, h_author, skip, 0, 1);
    int grid_p = cdiv(NP_ / 16, 8) * (DMODEL / 64);
    gemm_wmma_f16<<<grid_p, TB, 0, stream>>>(
        packT_p, packWa1, NP_ / 16, DMODEL, DMODEL, ba + DMODEL,
        out + (size_t)NA * DMODEL, h_paper, skip, 1, 1);
  }
}
